// MoELayer_24592982737070
// MI455X (gfx1250) — matrix operand order repaired and checked by reading, not compiled, not run on hardware
//
#include <hip/hip_runtime.h>
#include <math.h>

#define HDIM 1024
#define FDIM 4096
#define NEXP 32
#define TOPK 4
#define NTOK 2048
#define NPAIR (NTOK * TOPK)
#define ALPHA_LB 0.01f

#define BM 64
#define BN 64
#define KC 32

typedef __attribute__((ext_vector_type(16))) _Float16 v16h;
typedef __attribute__((ext_vector_type(8)))  float    v8f;

union ABFrag { uint4 u[2]; v16h v; };
union H8 { _Float16 h[8]; uint4 u; };

__device__ __forceinline__ float gelu_exact(float v) {
    return 0.5f * v * (1.0f + erff(v * 0.70710678118654752f));
}

// ---------------- 1. router: softmax + top-4 + counts ----------------
__global__ void router_kernel(const float* __restrict__ x,
                              const float* __restrict__ rw,
                              int* __restrict__ cnt,
                              int* __restrict__ topk_e,
                              float* __restrict__ topk_w)
{
    const int t = blockIdx.x;
    const int tid = threadIdx.x;            // 256 threads
    const int e = tid & (NEXP - 1);
    const int j = tid >> 5;                 // 8 partial groups
    const float* xr = x + (size_t)t * HDIM;
    float s = 0.f;
    for (int h = j; h < HDIM; h += 8)
        s += xr[h] * rw[h * NEXP + e];
    __shared__ float part[8][NEXP];
    part[j][e] = s;
    __syncthreads();
    if (tid < NEXP) {
        float l = 0.f;
        #pragma unroll
        for (int jj = 0; jj < 8; ++jj) l += part[jj][tid];
        part[0][tid] = l;
    }
    __syncthreads();
    if (tid == 0) {
        float pr[NEXP];
        float mx = -3.4e38f;
        for (int i = 0; i < NEXP; ++i) { pr[i] = part[0][i]; mx = fmaxf(mx, pr[i]); }
        float sum = 0.f;
        for (int i = 0; i < NEXP; ++i) { pr[i] = expf(pr[i] - mx); sum += pr[i]; }
        const float inv = 1.f / sum;
        for (int i = 0; i < NEXP; ++i) pr[i] *= inv;
        bool used[NEXP];
        for (int i = 0; i < NEXP; ++i) used[i] = false;
        int ti[TOPK]; float tp[TOPK]; float tsum = 0.f;
        for (int k = 0; k < TOPK; ++k) {
            int be = 0; float bv = -1.f;
            for (int i = 0; i < NEXP; ++i)
                if (!used[i] && pr[i] > bv) { bv = pr[i]; be = i; }
            used[be] = true; ti[k] = be; tp[k] = bv; tsum += bv;
        }
        const float itn = 1.f / tsum;
        for (int k = 0; k < TOPK; ++k) {
            topk_e[t * TOPK + k] = ti[k];
            topk_w[t * TOPK + k] = tp[k] * itn;
            atomicAdd(&cnt[ti[k]], 1);
        }
    }
}

// ---------------- 2. offsets + load-balance loss ----------------
__global__ void scan_kernel(const int* __restrict__ cnt,
                            int* __restrict__ off,
                            float* __restrict__ loss_out)
{
    if (threadIdx.x == 0 && blockIdx.x == 0) {
        int acc = 0, total = 0;
        for (int e = 0; e < NEXP; ++e) { off[e] = acc; acc += cnt[e]; total += cnt[e]; }
        off[NEXP] = acc;
        const float mean = (float)total / (float)(NTOK * NEXP);
        const float d = mean - 1.0f / (float)NEXP;
        *loss_out = ALPHA_LB * d * d;
    }
}

// ---------------- 3. scatter token->expert pair lists ----------------
__global__ void scatter_kernel(const int* __restrict__ topk_e,
                               const float* __restrict__ topk_w,
                               const int* __restrict__ off,
                               int* __restrict__ cursor,
                               int* __restrict__ pair_token,
                               int* __restrict__ pair_expert,
                               float* __restrict__ pair_w,
                               int* __restrict__ pair_pos)
{
    const int p = blockIdx.x * blockDim.x + threadIdx.x;
    if (p >= NPAIR) return;
    const int t = p >> 2;
    const int e = topk_e[p];
    const int pos = atomicAdd(&cursor[e], 1);
    const int g = off[e] + pos;
    pair_token[g]  = t;
    pair_expert[g] = e;
    pair_w[g]      = topk_w[p];
    pair_pos[p]    = g;     // inverse map for deterministic combine
}

// ---------------- 4. per-pair expert gate ----------------
__global__ void gate_kernel(const float* __restrict__ x,
                            const float* __restrict__ egw,
                            const int* __restrict__ pair_token,
                            const int* __restrict__ pair_expert,
                            const float* __restrict__ pair_w,
                            float* __restrict__ gate_pair)
{
    const int g = blockIdx.x;
    const int tid = threadIdx.x;
    const int t = pair_token[g];
    const int e = pair_expert[g];
    const float* xr = x + (size_t)t * HDIM;
    const float* gr = egw + (size_t)e * HDIM;
    float s = 0.f;
    for (int h = tid; h < HDIM; h += 256) s += xr[h] * gr[h];
    __shared__ float red[256];
    red[tid] = s;
    __syncthreads();
    for (int st = 128; st > 0; st >>= 1) {
        if (tid < st) red[tid] += red[tid + st];
        __syncthreads();
    }
    if (tid == 0)
        gate_pair[g] = pair_w[g] / (1.0f + expf(-red[0]));
}

// ---------------- 5. fc1: gather x, f32->f16, GEMM + GELU ----------------
__global__ __launch_bounds__(256)
void fc1_kernel(const float* __restrict__ x,
                const float* __restrict__ w1,
                const int* __restrict__ cnt,
                const int* __restrict__ off,
                const int* __restrict__ pair_token,
                _Float16* __restrict__ act)
{
    const int e = blockIdx.z;
    const int count = cnt[e];
    const int m0 = blockIdx.y * BM;
    if (m0 >= count) return;
    const int n0 = blockIdx.x * BN;
    const int base = off[e];
    const float* __restrict__ W = w1 + (size_t)e * HDIM * FDIM;

    __shared__ __align__(16) _Float16 As[BM][KC];     // row-major [m][k]
    __shared__ __align__(16) _Float16 Bs[BN][KC];     // transposed [n][k]
    __shared__ int toks[BM];

    const int tid = threadIdx.x;
    if (tid < BM) {
        const int gm = m0 + tid;
        toks[tid] = (gm < count) ? pair_token[base + gm] : -1;
    }

    const int ln = tid & 31, wv = tid >> 5;
    const int m_sub = wv & 3, n_half = wv >> 2;
    const int r = ln & 15, hf = ln >> 4;

    const int arow = tid >> 2, acol = (tid & 3) * 8;  // A stage: 64 rows x 32 k
    const int bn = tid & 63,   bkq = (tid >> 6) * 8;  // B stage: lane-contig n, 8 k each

    v8f c0 = {}, c1 = {};

    for (int k0 = 0; k0 < HDIM; k0 += KC) {
        __syncthreads();
        {   // A tile: gathered token rows, two b128 loads -> pack -> one b128 store
            const int tk = toks[arow];
            H8 pk;
            if (tk >= 0) {
                const float4* s4 = (const float4*)(x + (size_t)tk * HDIM + k0 + acol);
                const float4 f0 = s4[0], f1 = s4[1];
                pk.h[0] = (_Float16)f0.x; pk.h[1] = (_Float16)f0.y;
                pk.h[2] = (_Float16)f0.z; pk.h[3] = (_Float16)f0.w;
                pk.h[4] = (_Float16)f1.x; pk.h[5] = (_Float16)f1.y;
                pk.h[6] = (_Float16)f1.z; pk.h[7] = (_Float16)f1.w;
            } else {
                pk.u = make_uint4(0u, 0u, 0u, 0u);
            }
            *(uint4*)&As[arow][acol] = pk.u;
        }
        {   // B tile: k-strided loads (coalesced across lanes), one b128 store
            const float* src = W + (size_t)(k0 + bkq) * FDIM + n0 + bn;
            if (k0 + KC < HDIM) __builtin_prefetch(src + (size_t)KC * FDIM, 0, 1);
            H8 pk;
            #pragma unroll
            for (int q = 0; q < 8; ++q) pk.h[q] = (_Float16)src[(size_t)q * FDIM];
            *(uint4*)&Bs[bn][bkq] = pk.u;
        }
        __syncthreads();

        ABFrag a, b0f, b1f;
        const int rg = m_sub * 16 + r;
        a.u[0] = *(const uint4*)&As[rg][hf * 8];          // K 0-7 / 8-15
        a.u[1] = *(const uint4*)&As[rg][16 + hf * 8];     // K 16-23 / 24-31
        const int nA = n_half * 32 + r;
        const int kb = hf * 8;
        b0f.u[0] = *(const uint4*)&Bs[nA][kb];
        b0f.u[1] = *(const uint4*)&Bs[nA][kb + 16];
        b1f.u[0] = *(const uint4*)&Bs[nA + 16][kb];
        b1f.u[1] = *(const uint4*)&Bs[nA + 16][kb + 16];
        c0 = __builtin_amdgcn_wmma_f32_16x16x32_f16(false, a.v, false, b0f.v, (short)0, c0, false, false);
        c1 = __builtin_amdgcn_wmma_f32_16x16x32_f16(false, a.v, false, b1f.v, (short)0, c1, false, false);
    }

    const int mb = m_sub * 16 + hf * 8;
    const int nc = n_half * 32 + r;
    #pragma unroll
    for (int i = 0; i < 8; ++i) {
        const int gm = m0 + mb + i;
        if (gm < count) {
            _Float16* dst = act + (size_t)(base + gm) * FDIM + n0;
            dst[nc]      = (_Float16)gelu_exact(c0[i]);
            dst[nc + 16] = (_Float16)gelu_exact(c1[i]);
        }
    }
}

// ---------------- 6. fc2: async A staging, GEMM + gate scale ----------------
__global__ __launch_bounds__(256)
void fc2_kernel(const _Float16* __restrict__ act,
                const float* __restrict__ w2,
                const float* __restrict__ gate_pair,
                const int* __restrict__ cnt,
                const int* __restrict__ off,
                float* __restrict__ pairout)
{
    const int e = blockIdx.z;
    const int count = cnt[e];
    const int m0 = blockIdx.y * BM;
    if (m0 >= count) return;
    const int n0 = blockIdx.x * BN;
    const int base = off[e];
    const float* __restrict__ W = w2 + (size_t)e * FDIM * HDIM;

    __shared__ __align__(16) _Float16 As[BM][KC];
    __shared__ __align__(16) _Float16 Bs[BN][KC];

    const int tid = threadIdx.x;
    const int ln = tid & 31, wv = tid >> 5;
    const int m_sub = wv & 3, n_half = wv >> 2;
    const int r = ln & 15, hf = ln >> 4;

    const int arow = tid >> 2, acol = (tid & 3) * 8;
    const int bn = tid & 63,   bkq = (tid >> 6) * 8;

    // clamp OOB rows to a valid row: garbage only reaches C rows the
    // epilogue discards (WMMA rows are independent).
    int crow = m0 + arow; if (crow > count - 1) crow = count - 1;
    const _Float16* asrc = act + (size_t)(base + crow) * FDIM + acol;
    const unsigned lds_a = (unsigned)(uintptr_t)&As[arow][acol];

    v8f c0 = {}, c1 = {};

    for (int k0 = 0; k0 < FDIM; k0 += KC) {
        __syncthreads();
        {   // A tile: f16 activations, async DMA global -> LDS (no VGPR round trip)
            const unsigned long long ga = (unsigned long long)(uintptr_t)(asrc + k0);
            asm volatile("global_load_async_to_lds_b128 %0, %1, off"
                         :: "v"(lds_a), "v"(ga) : "memory");
        }
        {   // B tile: k-strided loads (coalesced across lanes), one b128 store
            const float* src = W + (size_t)(k0 + bkq) * HDIM + n0 + bn;
            if (k0 + KC < FDIM) __builtin_prefetch(src + (size_t)KC * HDIM, 0, 1);
            H8 pk;
            #pragma unroll
            for (int q = 0; q < 8; ++q) pk.h[q] = (_Float16)src[(size_t)q * HDIM];
            *(uint4*)&Bs[bn][bkq] = pk.u;
        }
        asm volatile("s_wait_asynccnt 0x0" ::: "memory");
        __syncthreads();

        ABFrag a, b0f, b1f;
        const int rg = m_sub * 16 + r;
        a.u[0] = *(const uint4*)&As[rg][hf * 8];
        a.u[1] = *(const uint4*)&As[rg][16 + hf * 8];
        const int nA = n_half * 32 + r;
        const int kb = hf * 8;
        b0f.u[0] = *(const uint4*)&Bs[nA][kb];
        b0f.u[1] = *(const uint4*)&Bs[nA][kb + 16];
        b1f.u[0] = *(const uint4*)&Bs[nA + 16][kb];
        b1f.u[1] = *(const uint4*)&Bs[nA + 16][kb + 16];
        c0 = __builtin_amdgcn_wmma_f32_16x16x32_f16(false, a.v, false, b0f.v, (short)0, c0, false, false);
        c1 = __builtin_amdgcn_wmma_f32_16x16x32_f16(false, a.v, false, b1f.v, (short)0, c1, false, false);
    }

    const int mb = m_sub * 16 + hf * 8;
    const int nc = n_half * 32 + r;
    #pragma unroll
    for (int i = 0; i < 8; ++i) {
        const int gm = m0 + mb + i;
        if (gm < count) {
            const float gte = gate_pair[base + gm];
            float* dst = pairout + (size_t)(base + gm) * HDIM + n0;
            dst[nc]      = c0[i] * gte;
            dst[nc + 16] = c1[i] * gte;
        }
    }
}

// ---------------- 7. deterministic combine (no float atomics) ----------------
__global__ void combine_kernel(const float* __restrict__ pairout,
                               const int* __restrict__ pair_pos,
                               float* __restrict__ out)
{
    const int idx = blockIdx.x * 256 + threadIdx.x;   // t*H + h
    if (idx >= NTOK * HDIM) return;
    const int t = idx >> 10;
    const int h = idx & (HDIM - 1);
    float s = 0.f;
    #pragma unroll
    for (int k = 0; k < TOPK; ++k) {
        const int g = pair_pos[t * TOPK + k];
        s += pairout[(size_t)g * HDIM + h];
    }
    out[idx] = s;
}

extern "C" void kernel_launch(void* const* d_in, const int* in_sizes, int n_in,
                              void* d_out, int out_size, void* d_ws, size_t ws_size,
                              hipStream_t stream)
{
    (void)in_sizes; (void)n_in; (void)out_size; (void)ws_size;
    const float* x   = (const float*)d_in[0];
    const float* rw  = (const float*)d_in[1];
    const float* egw = (const float*)d_in[2];
    const float* w1  = (const float*)d_in[3];
    const float* w2  = (const float*)d_in[4];
    float* out = (float*)d_out;

    size_t o = 0;
    auto take = [&](size_t n, size_t al) {
        o = (o + al - 1) & ~(al - 1);
        size_t r = o; o += n; return r;
    };
    char* ws = (char*)d_ws;
    int*      cnt         = (int*)   (ws + take(NEXP * 4, 16));
    int*      cursor      = (int*)   (ws + take(NEXP * 4, 16));
    int*      off         = (int*)   (ws + take((NEXP + 1) * 4, 16));
    int*      topk_e      = (int*)   (ws + take((size_t)NPAIR * 4, 16));
    float*    topk_w      = (float*) (ws + take((size_t)NPAIR * 4, 16));
    int*      pair_token  = (int*)   (ws + take((size_t)NPAIR * 4, 16));
    int*      pair_expert = (int*)   (ws + take((size_t)NPAIR * 4, 16));
    float*    pair_w      = (float*) (ws + take((size_t)NPAIR * 4, 16));
    int*      pair_pos    = (int*)   (ws + take((size_t)NPAIR * 4, 16));
    float*    gate_p      = (float*) (ws + take((size_t)NPAIR * 4, 16));
    float*    pairout     = (float*) (ws + take((size_t)NPAIR * HDIM * 4, 256));
    _Float16* actb        = (_Float16*)(ws + take((size_t)NPAIR * FDIM * 2, 256));

    hipMemsetAsync(cnt, 0, NEXP * 4, stream);
    hipMemsetAsync(cursor, 0, NEXP * 4, stream);

    router_kernel<<<NTOK, 256, 0, stream>>>(x, rw, cnt, topk_e, topk_w);
    scan_kernel<<<1, 32, 0, stream>>>(cnt, off, out + (size_t)NTOK * HDIM);
    scatter_kernel<<<(NPAIR + 255) / 256, 256, 0, stream>>>(
        topk_e, topk_w, off, cursor, pair_token, pair_expert, pair_w, pair_pos);
    gate_kernel<<<NPAIR, 256, 0, stream>>>(x, egw, pair_token, pair_expert, pair_w, gate_p);
    fc1_kernel<<<dim3(FDIM / BN, (NTOK + BM - 1) / BM, NEXP), 256, 0, stream>>>(
        x, w1, cnt, off, pair_token, actb);
    fc2_kernel<<<dim3(HDIM / BN, (NTOK + BM - 1) / BM, NEXP), 256, 0, stream>>>(
        actb, w2, gate_p, cnt, off, pairout);
    combine_kernel<<<(NTOK * HDIM + 255) / 256, 256, 0, stream>>>(pairout, pair_pos, out);
}
